// SelfAttentionLayer_420906795813
// MI455X (gfx1250) — matrix operand order repaired and checked by reading, not compiled, not run on hardware
//
#include <hip/hip_runtime.h>

#define BATCH 4
#define SLEN  4096
#define DDIM  256

typedef __bf16 bf16_t;
typedef bf16_t v16bf __attribute__((ext_vector_type(16)));
typedef float  v8f   __attribute__((ext_vector_type(8)));
typedef float  v2f   __attribute__((ext_vector_type(2)));

union ABfrag {
  uint4  u[2];   // 8 VGPRs = 16 bf16 values per lane
  v16bf  v;
};

// round-to-nearest-even f32 -> bf16 bits
__device__ __forceinline__ unsigned short f2bf(float f) {
  unsigned u = __float_as_uint(f);
  u += 0x7fffu + ((u >> 16) & 1u);
  return (unsigned short)(u >> 16);
}

__device__ __forceinline__ v8f wmma_bf16(const ABfrag& a, const ABfrag& b, v8f c) {
  return __builtin_amdgcn_wmma_f32_16x16x32_bf16(false, a.v, false, b.v,
                                                 (short)0, c, false, false);
}

// CDNA5 async DMA: global -> LDS, 16B per lane, tracked by ASYNCcnt.
// GV mode: vdst = LDS byte offset (low 32 bits of generic shared pointer),
//          vaddr = 64-bit global address in a VGPR pair.
__device__ __forceinline__ void async_b128(const void* g, const void* l) {
  unsigned lds = (unsigned)(uintptr_t)l;
  unsigned long long ga = (unsigned long long)(uintptr_t)g;
  asm volatile("global_load_async_to_lds_b128 %0, %1, off"
               :: "v"(lds), "v"(ga) : "memory");
}
__device__ __forceinline__ void wait_async0() {
  asm volatile("s_wait_asynccnt 0x0" ::: "memory");
}

// ---------------------------------------------------------------------------
// Kernel 1: q = x@Wq^T + bq (scaled by 1/4 -> folds 1/sqrt(D) into scores),
//           v = x@Wv^T + bv (stored transposed [D,S] as bf16).
// One block per 16-row s-tile; x tile staged via async DMA; f32 WMMA 16x16x4.
// ---------------------------------------------------------------------------
__global__ __launch_bounds__(128) void qv_proj_kernel(
    const float* __restrict__ x, const float* __restrict__ Wq,
    const float* __restrict__ bq, const float* __restrict__ Wv,
    const float* __restrict__ bv, unsigned short* __restrict__ qbf,
    unsigned short* __restrict__ vtbf) {
  __shared__ __align__(16) float xs[16][260];   // pitch 1040B: conflict-free b64 reads
  const int tid  = threadIdx.x;
  const int b    = blockIdx.x >> 8;             // 256 s-tiles per batch
  const int srow = (blockIdx.x & 255) << 4;

  const float* xg = x + ((size_t)b * SLEN + srow) * DDIM;
#pragma unroll
  for (int i = 0; i < 8; ++i) {                 // 16x256 f32 = 1024 b128 DMAs
    int idx = tid + i * 128;
    int r = idx >> 6, c4 = idx & 63;
    async_b128(xg + r * DDIM + c4 * 4, &xs[r][c4 * 4]);
  }
  wait_async0();
  __syncthreads();

  const int lane = tid & 31, wv = tid >> 5;
  const int mn = lane & 15, half = lane >> 4;   // A row m == B col n == lane%16
#pragma unroll
  for (int eti = 0; eti < 4; ++eti) {
    const int e = ((wv << 2) + eti) * 16 + mn;
    const float* wqr = Wq + (size_t)e * DDIM;
    const float* wvr = Wv + (size_t)e * DDIM;
    v8f aq = {0,0,0,0,0,0,0,0};
    v8f av = {0,0,0,0,0,0,0,0};
#pragma unroll 8
    for (int kc = 0; kc < 64; ++kc) {           // K = 256 in steps of 4
      int k0 = kc * 4 + half * 2;
      v2f A  = *(const v2f*)(&xs[mn][k0]);      // x[srow+m][k0..k0+1]
      v2f Bq = *(const v2f*)(wqr + k0);         // Wq[e][k0..k0+1]
      v2f Bv = *(const v2f*)(wvr + k0);
      aq = __builtin_amdgcn_wmma_f32_16x16x4_f32(false, A, false, Bq, (short)0, aq, false, false);
      av = __builtin_amdgcn_wmma_f32_16x16x4_f32(false, A, false, Bv, (short)0, av, false, false);
    }
    const float biq = bq[e], biv = bv[e];
    // Q store (row-major bf16), scaled 0.25 so q'.q' = q.q / sqrt(256)
#pragma unroll
    for (int g = 0; g < 8; ++g) {
      int srw = srow + g + 8 * half;            // C/D layout: M = g + 8*(lane/16)
      qbf[((size_t)b * SLEN + srw) * DDIM + e] = f2bf(0.25f * (aq[g] + biq));
    }
    // V^T store: lane owns column e, rows srow+8*half .. +7 contiguous -> one b128
    uint4 pk; unsigned lo, hi;
    lo = f2bf(av[0] + biv); hi = f2bf(av[1] + biv); pk.x = lo | (hi << 16);
    lo = f2bf(av[2] + biv); hi = f2bf(av[3] + biv); pk.y = lo | (hi << 16);
    lo = f2bf(av[4] + biv); hi = f2bf(av[5] + biv); pk.z = lo | (hi << 16);
    lo = f2bf(av[6] + biv); hi = f2bf(av[7] + biv); pk.w = lo | (hi << 16);
    *(uint4*)(vtbf + ((size_t)b * DDIM + e) * SLEN + srow + 8 * half) = pk;
  }
}

// ---------------------------------------------------------------------------
// Kernel 2: flash attention. scores = qs@qs^T, online softmax, O = P@V.
// 4 waves/block; wave owns 16 query rows (Q frags in VGPRs, O in 128 VGPRs).
// Double-buffered LDS chunks filled by async DMA, overlapped with WMMA;
// one barrier per 32-key step.
// ---------------------------------------------------------------------------
__global__ __launch_bounds__(128) void flash_attn_kernel(
    const unsigned short* __restrict__ q, const unsigned short* __restrict__ vt,
    float* __restrict__ out) {
  __shared__ __align__(16) unsigned short qj [2][32 * 264];    // 2 x 16896B
  __shared__ __align__(16) unsigned short vts[2][256 * 40];    // 2 x 20480B
  __shared__ __align__(16) unsigned short pls[4 * 16 * 40];    // per-wave P bounce

  const int tid   = threadIdx.x;
  const int b     = blockIdx.x >> 6;            // 64 query blocks per batch
  const int srow0 = (blockIdx.x & 63) << 6;
  const int lane  = tid & 31, wv = tid >> 5;
  const int n = lane & 15, half = lane >> 4;
  const int qs = srow0 + wv * 16;

  const unsigned short* qgb = q  + (size_t)b * SLEN * DDIM;
  const unsigned short* vgb = vt + (size_t)b * DDIM * SLEN;

  // per-thread share of one 32-key chunk: 8 b128 of Qj + 8 b128 of V^T
  const int qr = tid >> 2, qc = tid & 3;            // covers 128 of 1024 each iter
  // resident A-fragments: this wave's 16 query rows x 256 dims (bf16)
  uint4 qi[8][2];
  const unsigned short* qrow = qgb + (size_t)(qs + n) * DDIM;
#pragma unroll
  for (int kc = 0; kc < 8; ++kc) {
    qi[kc][0] = *(const uint4*)(qrow + kc * 32 + half * 8);       // K=h*8..+7
    qi[kc][1] = *(const uint4*)(qrow + kc * 32 + 16 + half * 8);  // K=16+h*8..+7
  }

  v8f zero = {0,0,0,0,0,0,0,0};
  v8f o[16];
#pragma unroll
  for (int nt = 0; nt < 16; ++nt) o[nt] = zero;
  float mst[8], lst[8];
#pragma unroll
  for (int g = 0; g < 8; ++g) { mst[g] = -3.0e38f; lst[g] = 0.0f; }

  unsigned short* pw = pls + wv * (16 * 40);
  const int NJ = SLEN / 32;

  // prologue: DMA chunk 0 into buffer 0
#pragma unroll
  for (int i = 0; i < 8; ++i) {
    int idx = tid + i * 128;
    { int r = idx >> 5, c = idx & 31;
      async_b128(qgb + (size_t)r * DDIM + c * 8, &qj[0][r * 264 + c * 8]); }
    { int r = idx >> 2, c = idx & 3;
      async_b128(vgb + (size_t)r * SLEN + c * 8, &vts[0][r * 40 + c * 8]); }
  }
  wait_async0();
  __syncthreads();

  for (int jc = 0; jc < NJ; ++jc) {
    const int cur = jc & 1;
    if (jc + 1 < NJ) {                 // DMA next chunk while computing this one
      const int j1 = (jc + 1) * 32;
      unsigned short* qjn = qj[cur ^ 1];
      unsigned short* vtn = vts[cur ^ 1];
#pragma unroll
      for (int i = 0; i < 8; ++i) {
        int idx = tid + i * 128;
        { int r = idx >> 5, c = idx & 31;
          async_b128(qgb + (size_t)(j1 + r) * DDIM + c * 8, qjn + r * 264 + c * 8); }
        { int r = idx >> 2, c = idx & 3;
          async_b128(vgb + (size_t)r * SLEN + j1 + c * 8, vtn + r * 40 + c * 8); }
      }
    }
    const unsigned short* qjc = qj[cur];
    const unsigned short* vtc = vts[cur];

    // S[16x32] = Qi x Qj^T, f32 accumulate
    v8f s0 = zero, s1 = zero;
#pragma unroll
    for (int kc = 0; kc < 8; ++kc) {
      ABfrag a; a.u[0] = qi[kc][0]; a.u[1] = qi[kc][1];
      ABfrag b0, b1;
      const unsigned short* p0 = qjc + n * 264 + kc * 32 + half * 8;
      b0.u[0] = *(const uint4*)(p0);      b0.u[1] = *(const uint4*)(p0 + 16);
      const unsigned short* p1 = qjc + (n + 16) * 264 + kc * 32 + half * 8;
      b1.u[0] = *(const uint4*)(p1);      b1.u[1] = *(const uint4*)(p1 + 16);
      s0 = wmma_bf16(a, b0, s0);
      s1 = wmma_bf16(a, b1, s1);
    }

    // online softmax across the 32 new columns; row g+8*half lives in one
    // 16-lane half-group, xor masks 1/2/4/8 reduce within it
    float alpha[8];
#pragma unroll
    for (int g = 0; g < 8; ++g) {
      float v = fmaxf(s0[g], s1[g]);
      v = fmaxf(v, __shfl_xor(v, 1, 32));
      v = fmaxf(v, __shfl_xor(v, 2, 32));
      v = fmaxf(v, __shfl_xor(v, 4, 32));
      v = fmaxf(v, __shfl_xor(v, 8, 32));
      float mnew = fmaxf(mst[g], v);
      alpha[g] = __expf(mst[g] - mnew);
      float p0 = __expf(s0[g] - mnew);
      float p1 = __expf(s1[g] - mnew);
      mst[g] = mnew;
      float rs = p0 + p1;
      rs += __shfl_xor(rs, 1, 32);
      rs += __shfl_xor(rs, 2, 32);
      rs += __shfl_xor(rs, 4, 32);
      rs += __shfl_xor(rs, 8, 32);
      lst[g] = lst[g] * alpha[g] + rs;
      const int M = g + 8 * half;               // C-layout row of this element
      pw[M * 40 + n]      = f2bf(p0);
      pw[M * 40 + 16 + n] = f2bf(p1);
    }

    // reload P as an A-fragment (C-layout -> A-layout via per-wave LDS bounce)
    ABfrag pa;
    const unsigned short* pr = pw + n * 40;     // A row m = lane%16
    pa.u[0] = *(const uint4*)(pr + half * 8);
    pa.u[1] = *(const uint4*)(pr + 16 + half * 8);

    // O = O*alpha + P x V^T-chunk  (16 output N-tiles of 16 dims)
#pragma unroll
    for (int nt = 0; nt < 16; ++nt) {
      v8f acc = o[nt];
#pragma unroll
      for (int g = 0; g < 8; ++g) acc[g] *= alpha[g];
      ABfrag bvf;
      const unsigned short* vp = vtc + (nt * 16 + n) * 40 + half * 8;
      bvf.u[0] = *(const uint4*)(vp);
      bvf.u[1] = *(const uint4*)(vp + 16);
      o[nt] = wmma_bf16(pa, bvf, acc);
    }

    wait_async0();      // own next-chunk DMAs done (no-op on last iteration)
    __syncthreads();    // publish DMAs; all readers of buf[cur] finished
  }

  float inv[8];
#pragma unroll
  for (int g = 0; g < 8; ++g) inv[g] = 1.0f / lst[g];
  float* ob = out + ((size_t)b * SLEN + qs) * DDIM;
#pragma unroll
  for (int nt = 0; nt < 16; ++nt)
#pragma unroll
    for (int g = 0; g < 8; ++g)
      ob[(size_t)(g + 8 * half) * DDIM + nt * 16 + n] = o[nt][g] * inv[g];
}

// ---------------------------------------------------------------------------
extern "C" void kernel_launch(void* const* d_in, const int* in_sizes, int n_in,
                              void* d_out, int out_size, void* d_ws, size_t ws_size,
                              hipStream_t stream) {
  const float* x  = (const float*)d_in[0];
  const float* Wq = (const float*)d_in[1];
  const float* bq = (const float*)d_in[2];
  // d_in[3] = Wk, d_in[4] = bk : computed-but-unused in the reference -> skipped
  const float* Wv = (const float*)d_in[5];
  const float* bv = (const float*)d_in[6];

  unsigned short* qbf  = (unsigned short*)d_ws;                        // 8 MB
  unsigned short* vtbf = qbf + (size_t)BATCH * SLEN * DDIM;            // 8 MB

  qv_proj_kernel<<<BATCH * (SLEN / 16), 128, 0, stream>>>(x, Wq, bq, Wv, bv, qbf, vtbf);
  flash_attn_kernel<<<BATCH * (SLEN / 64), 128, 0, stream>>>(qbf, vtbf, (float*)d_out);
}
